// Attention_59631325937791
// MI455X (gfx1250) — matrix operand order repaired and checked by reading, not compiled, not run on hardware
//
#include <hip/hip_runtime.h>
#include <hip/hip_bf16.h>

typedef __attribute__((ext_vector_type(16))) _Float16 v16h;
typedef __attribute__((ext_vector_type(8)))  _Float16 v8h;
typedef __attribute__((ext_vector_type(8)))  float    v8f;
typedef __attribute__((ext_vector_type(4)))  float    v4f;

#define DEV static __device__ __forceinline__

DEV v16h cat8(v8h lo, v8h hi) {
  return __builtin_shufflevector(lo, hi, 0,1,2,3,4,5,6,7,8,9,10,11,12,13,14,15);
}
DEV v8f wmma_f16(v16h a, v16h b, v8f c) {
  // D = A*B + C, 16x16x32, f16 in, f32 acc
  return __builtin_amdgcn_wmma_f32_16x16x32_f16(false, a, false, b, (short)0, c, false, false);
}

// CDNA5 async DMA: global -> LDS without VGPR transit, tracked by ASYNCcnt.
DEV void async_copy_b128(const _Float16* g, _Float16* l) {
  unsigned loff = (unsigned)(uintptr_t)l;   // low 32 bits of generic ptr = LDS offset
  asm volatile("global_load_async_to_lds_b128 %0, %1, off"
               :: "v"(loff), "v"((unsigned long long)(uintptr_t)g)
               : "memory");
}
DEV void async_wait0() { asm volatile("s_wait_asynccnt 0x0" ::: "memory"); }

// ---------------------------------------------------------------------------
// Prep kernels
// ---------------------------------------------------------------------------
__global__ void f32_to_f16_kernel(const float* __restrict__ in, _Float16* __restrict__ out, int n) {
  int i = blockIdx.x * 256 + threadIdx.x;
  if (i < n) out[i] = (_Float16)in[i];
}

// in: [R][C] f32 row-major -> out: [C][R] f16 (i.e. out[n][k] = in[k][n])
__global__ void transpose_to_f16_kernel(const float* __restrict__ in, _Float16* __restrict__ out,
                                        int R, int C) {
  int i = blockIdx.x * 256 + threadIdx.x;
  if (i < R * C) {
    int c = i / R;
    int r = i - c * R;
    out[i] = (_Float16)in[(size_t)r * C + c];
  }
}

__global__ void mask2_kernel(const float* __restrict__ m, float* __restrict__ out, int n) {
  int i = blockIdx.x * 256 + threadIdx.x;
  if (i < n) out[i] = 2.0f * m[i];   // reference adds the mask twice
}

// ---------------------------------------------------------------------------
// WMMA GEMM:  C[M][N] = A[M][K] * Bt[N][K]^T + bias[N]
// block tile 128x128, BK=32, 256 threads = 8 waves, wave tile 32x64 (2x4 WMMA)
// Double-buffered LDS filled by async global->LDS DMA; one barrier / K-step.
// ---------------------------------------------------------------------------
template <typename OutT>
__global__ __launch_bounds__(256)
void gemm_kernel(const _Float16* __restrict__ A, const _Float16* __restrict__ Bt,
                 const float* __restrict__ bias, OutT* __restrict__ C,
                 int M, int N, int K)
{
  __shared__ _Float16 lA[2][128][40];   // +8 pad: row stride 80B (16B aligned)
  __shared__ _Float16 lB[2][128][40];

  const int tid  = threadIdx.x;
  const int lane = tid & 31;
  const int wave = tid >> 5;      // 0..7
  const int wm   = wave >> 1;     // 0..3 (M dir)
  const int wn   = wave & 1;      // 0..1 (N dir)
  const int half = lane >> 4;     // 0/1
  const int l15  = lane & 15;
  const int m0   = blockIdx.y * 128;
  const int n0   = blockIdx.x * 128;

  // staging coords: each thread owns 16B chunks at (rowS, chS) and (rowS+64, chS)
  const int rowS = tid >> 2;
  const int chS  = (tid & 3) * 8;

  auto stage = [&](int buf, int kb) {
    async_copy_b128(&A[(size_t)(m0 + rowS) * K + kb + chS],      &lA[buf][rowS][chS]);
    async_copy_b128(&A[(size_t)(m0 + rowS + 64) * K + kb + chS], &lA[buf][rowS + 64][chS]);
    async_copy_b128(&Bt[(size_t)(n0 + rowS) * K + kb + chS],      &lB[buf][rowS][chS]);
    async_copy_b128(&Bt[(size_t)(n0 + rowS + 64) * K + kb + chS], &lB[buf][rowS + 64][chS]);
  };

  const v8f Z = {0.f,0.f,0.f,0.f,0.f,0.f,0.f,0.f};
  v8f acc[2][4];
#pragma unroll
  for (int i = 0; i < 2; ++i)
#pragma unroll
    for (int j = 0; j < 4; ++j) acc[i][j] = Z;

  stage(0, 0);
  async_wait0();
  __syncthreads();

  int cur = 0;
  for (int kb = 0; kb < K; kb += 32) {
    if (kb + 32 < K) stage(cur ^ 1, kb + 32);   // overlap DMA with WMMA

    v16h aF[2], bF[4];
#pragma unroll
    for (int i = 0; i < 2; ++i) {
      int r = wm * 32 + i * 16 + l15;
      v8h lo = *(const v8h*)&lA[cur][r][half * 8];
      v8h hi = *(const v8h*)&lA[cur][r][16 + half * 8];
      aF[i] = cat8(lo, hi);
    }
#pragma unroll
    for (int j = 0; j < 4; ++j) {
      int n = wn * 64 + j * 16 + l15;
      v8h lo = *(const v8h*)&lB[cur][n][half * 8];
      v8h hi = *(const v8h*)&lB[cur][n][16 + half * 8];
      bF[j] = cat8(lo, hi);
    }
#pragma unroll
    for (int i = 0; i < 2; ++i)
#pragma unroll
      for (int j = 0; j < 4; ++j)
        acc[i][j] = wmma_f16(aF[i], bF[j], acc[i][j]);

    async_wait0();
    __syncthreads();
    cur ^= 1;
  }

  // epilogue: + bias, store
#pragma unroll
  for (int i = 0; i < 2; ++i)
#pragma unroll
    for (int j = 0; j < 4; ++j) {
      int col = n0 + wn * 64 + j * 16 + l15;
      float bv = bias[col];
#pragma unroll
      for (int r = 0; r < 8; ++r) {
        int row = m0 + wm * 32 + i * 16 + r + half * 8;
        C[(size_t)row * N + col] = (OutT)(acc[i][j][r] + bv);
      }
    }
}

// ---------------------------------------------------------------------------
// Repack qkv[4096][3072] -> Qs/Kf [BH][T][64] (Q pre-scaled 1/sqrt(64)),
// Vt [BH][64][T]
// ---------------------------------------------------------------------------
__global__ void repack_kernel(const _Float16* __restrict__ qkv,
                              _Float16* __restrict__ Qs, _Float16* __restrict__ Kf,
                              _Float16* __restrict__ Vt)
{
  int idx = blockIdx.x * 256 + threadIdx.x;       // 32*2048*64 = 4194304
  if (idx >= 32 * 2048 * 64) return;
  int d  = idx & 63;
  int t  = (idx >> 6) & 2047;
  int bh = idx >> 17;
  int b = bh >> 4, h = bh & 15;
  const _Float16* row = qkv + (size_t)(b * 2048 + t) * 3072 + h * 64 + d;
  Qs[idx] = (_Float16)((float)row[0] * 0.125f);
  Kf[idx] = row[1024];
  Vt[((size_t)bh * 64 + d) * 2048 + t] = row[2048];
}

// ---------------------------------------------------------------------------
// Flash attention. grid = (T/128, B*H), 256 thr = 8 waves, wave owns 16 Q rows.
// Computes S^T = K * Q^T so the P fragment for O = P*V needs no LDS transpose.
// K/V tiles double-buffered in LDS via async global->LDS DMA.
// ---------------------------------------------------------------------------
__global__ __launch_bounds__(256)
void attn_kernel(const _Float16* __restrict__ Qs,   // [32][2048][64] (pre-scaled)
                 const _Float16* __restrict__ Kf,   // [32][2048][64]
                 const _Float16* __restrict__ Vt,   // [32][64][2048]
                 const float* __restrict__ mask2,   // [B][2048], already doubled
                 _Float16* __restrict__ Oh)         // [4096][1024] f16
{
  __shared__ _Float16 lK[2][64][72];   // key rows x hd     (row stride 144B)
  __shared__ _Float16 lV[2][64][72];   // hd rows x keys
  const int tid  = threadIdx.x;
  const int lane = tid & 31;
  const int wave = tid >> 5;
  const int half = lane >> 4;
  const int l15  = lane & 15;
  const int bh   = blockIdx.y;
  const int b    = bh >> 4;
  const int h    = bh & 15;
  const int f0   = blockIdx.x * 128 + wave * 16;   // Q-row base (within T)

  const _Float16* Qbh = Qs + (size_t)bh * 2048 * 64;
  const _Float16* Kbh = Kf + (size_t)bh * 2048 * 64;
  const _Float16* Vbh = Vt + (size_t)bh * 64 * 2048;
  const float*    mb  = mask2 + (size_t)b * 2048;

  // staging coords: chunks at (rowA, chA) and (rowA+32, chA)
  const int rowA = tid >> 3;           // 0..31
  const int chA  = (tid & 7) * 8;      // 0..56

  auto stage = [&](int buf, int kb) {
    async_copy_b128(&Kbh[(size_t)(kb + rowA) * 64 + chA],      &lK[buf][rowA][chA]);
    async_copy_b128(&Kbh[(size_t)(kb + rowA + 32) * 64 + chA], &lK[buf][rowA + 32][chA]);
    async_copy_b128(&Vbh[(size_t)rowA * 2048 + kb + chA],        &lV[buf][rowA][chA]);
    async_copy_b128(&Vbh[(size_t)(rowA + 32) * 2048 + kb + chA], &lV[buf][rowA + 32][chA]);
  };

  // Q as B-operand of S^T = K*Q^T : lane = Q row, K-dim = hd
  v16h qb[2];
#pragma unroll
  for (int s = 0; s < 2; ++s) {
    const _Float16* qp = &Qbh[(size_t)(f0 + l15) * 64 + s * 32 + half * 8];
    qb[s] = cat8(*(const v8h*)qp, *(const v8h*)(qp + 16));
  }

  const v8f Z = {0.f,0.f,0.f,0.f,0.f,0.f,0.f,0.f};
  v8f o[4];
#pragma unroll
  for (int j = 0; j < 4; ++j) o[j] = Z;
  float row_m = -INFINITY, row_l = 0.f;

  stage(0, 0);
  async_wait0();
  __syncthreads();

  int cur = 0;
  for (int kb = 0; kb < 2048; kb += 64) {
    if (kb + 64 < 2048) stage(cur ^ 1, kb + 64);   // overlap DMA with compute

    // S^T tiles: st[kt] is [16 keys][16 Q-rows]; lane = Q row, slot = key
    v8f st[4];
#pragma unroll
    for (int kt = 0; kt < 4; ++kt) {
      v8f a = Z;
#pragma unroll
      for (int s = 0; s < 2; ++s) {
        int krow = kt * 16 + l15;
        v8h lo = *(const v8h*)&lK[cur][krow][s * 32 + half * 8];
        v8h hi = *(const v8h*)&lK[cur][krow][s * 32 + 16 + half * 8];
        a = wmma_f16(cat8(lo, hi), qb[s], a);
      }
      // + 2*mask[key]; slot r <-> key = kb + kt*16 + half*8 + r
      const v4f* mp = (const v4f*)&mb[kb + kt * 16 + half * 8];
      v4f mv0 = mp[0], mv1 = mp[1];
#pragma unroll
      for (int r = 0; r < 4; ++r) { a[r] += mv0[r]; a[r + 4] += mv1[r]; }
      st[kt] = a;
    }

    // online softmax along keys (the M dim of st): lane pair (l, l+16) owns one row
    float bmax = -INFINITY;
#pragma unroll
    for (int kt = 0; kt < 4; ++kt)
#pragma unroll
      for (int r = 0; r < 8; ++r) bmax = fmaxf(bmax, st[kt][r]);
    bmax = fmaxf(bmax, __shfl_xor(bmax, 16, 32));
    float newm = fmaxf(row_m, bmax);
    float corr = __expf(row_m - newm);
    row_m = newm;
    float psum = 0.f;
#pragma unroll
    for (int kt = 0; kt < 4; ++kt)
#pragma unroll
      for (int r = 0; r < 8; ++r) {
        float p = __expf(st[kt][r] - newm);
        st[kt][r] = p;
        psum += p;
      }
    psum += __shfl_xor(psum, 16, 32);
    row_l = row_l * corr + psum;

    // rescale O: row r of O tile is Q-row (r + half*8); its corr lives in that lane
    float cr[8];
#pragma unroll
    for (int r = 0; r < 8; ++r) cr[r] = __shfl(corr, r + half * 8, 32);
#pragma unroll
    for (int j = 0; j < 4; ++j)
#pragma unroll
      for (int r = 0; r < 8; ++r) o[j][r] *= cr[r];

    // O += P * V ; P's A-fragment falls straight out of st register layout
#pragma unroll
    for (int s = 0; s < 2; ++s) {
      v16h pa;
#pragma unroll
      for (int r = 0; r < 8; ++r) {
        pa[r]     = (_Float16)st[2 * s][r];
        pa[r + 8] = (_Float16)st[2 * s + 1][r];
      }
#pragma unroll
      for (int j = 0; j < 4; ++j) {
        int n = j * 16 + l15;
        v8h lo = *(const v8h*)&lV[cur][n][s * 32 + half * 8];
        v8h hi = *(const v8h*)&lV[cur][n][s * 32 + 16 + half * 8];
        o[j] = wmma_f16(pa, cat8(lo, hi), o[j]);
      }
    }

    async_wait0();
    __syncthreads();
    cur ^= 1;
  }

  // normalize by row sums and store f16 attn output [token][D]
  float linv = 1.0f / row_l;
  float lr[8];
#pragma unroll
  for (int r = 0; r < 8; ++r) lr[r] = __shfl(linv, r + half * 8, 32);
  const int tokBase = b * 2048 + blockIdx.x * 128 + wave * 16;
#pragma unroll
  for (int j = 0; j < 4; ++j) {
    int col = h * 64 + j * 16 + l15;
#pragma unroll
    for (int r = 0; r < 8; ++r) {
      int tok = tokBase + r + half * 8;
      Oh[(size_t)tok * 1024 + col] = (_Float16)(o[j][r] * lr[r]);
    }
  }
}

// ---------------------------------------------------------------------------
extern "C" void kernel_launch(void* const* d_in, const int* in_sizes, int n_in,
                              void* d_out, int out_size, void* d_ws, size_t ws_size,
                              hipStream_t stream) {
  const float* x      = (const float*)d_in[0];
  const float* amask  = (const float*)d_in[1];
  const float* w_attn = (const float*)d_in[2];
  const float* b_attn = (const float*)d_in[3];
  const float* w_proj = (const float*)d_in[4];
  const float* b_proj = (const float*)d_in[5];
  float* out = (float*)d_out;

  char* ws = (char*)d_ws;
  size_t off = 0;
  auto take = [&](size_t bytes) {
    char* p = ws + off;
    off = (off + bytes + 255) & ~(size_t)255;
    return p;
  };
  _Float16* xf  = (_Float16*)take((size_t)4096 * 1024 * 2);  // x as f16
  _Float16* wTa = (_Float16*)take((size_t)3072 * 1024 * 2);  // w_attn^T f16
  _Float16* wTp = (_Float16*)take((size_t)1024 * 1024 * 2);  // w_proj^T f16
  _Float16* qkv = (_Float16*)take((size_t)4096 * 3072 * 2);  // qkv f16
  _Float16* Qs  = (_Float16*)take((size_t)32 * 2048 * 64 * 2);
  _Float16* Kf  = (_Float16*)take((size_t)32 * 2048 * 64 * 2);
  _Float16* Vt  = (_Float16*)take((size_t)32 * 64 * 2048 * 2);
  _Float16* Of  = (_Float16*)take((size_t)4096 * 1024 * 2);  // attn out f16
  float*    m2  = (float*)take((size_t)4096 * 4);            // doubled mask

  f32_to_f16_kernel<<<16384, 256, 0, stream>>>(x, xf, 4096 * 1024);
  transpose_to_f16_kernel<<<12288, 256, 0, stream>>>(w_attn, wTa, 1024, 3072);
  transpose_to_f16_kernel<<<4096, 256, 0, stream>>>(w_proj, wTp, 1024, 1024);
  mask2_kernel<<<16, 256, 0, stream>>>(amask, m2, 4096);

  // qkv = x @ w_attn + b_attn
  gemm_kernel<_Float16><<<dim3(24, 32), 256, 0, stream>>>(xf, wTa, b_attn, qkv,
                                                          4096, 3072, 1024);
  repack_kernel<<<16384, 256, 0, stream>>>(qkv, Qs, Kf, Vt);
  attn_kernel<<<dim3(16, 32), 256, 0, stream>>>(Qs, Kf, Vt, m2, Of);
  // out = attn @ w_proj + b_proj
  gemm_kernel<float><<<dim3(8, 32), 256, 0, stream>>>(Of, wTp, b_proj, out,
                                                      4096, 1024, 1024);
}
